// Contrast_loss_point_cloud_4690104287785
// MI455X (gfx1250) — matrix operand order repaired and checked by reading, not compiled, not run on hardware
//
#include <hip/hip_runtime.h>

#define B_   8
#define C_   128
#define N_   4096
#define NCLS 16

#define JB          64              // columns staged per LDS block
#define LDS_STRIDE  (C_ + 8)        // halves; 272B = 68 dwords -> 4-bank skew/col
#define JSPLIT      2               // j-sweep split across workgroups
#define JLEN        (N_ / JSPLIT)   // 2048 columns per workgroup
#define NBLK        (JLEN / JB)     // 32 staged blocks per workgroup

typedef _Float16 half_t;
typedef __attribute__((ext_vector_type(16))) _Float16 v16h;
typedef __attribute__((ext_vector_type(8)))  _Float16 v8h;
typedef __attribute__((ext_vector_type(8)))  float    v8f;

// ---------------------------------------------------------------------------
// Kernel 1: per-point L2 normalize + fp32 -> fp16 convert.
// features layout: [B, C, N]  ->  Vh layout: [B, N, C] (row-major, contiguous C)
// ---------------------------------------------------------------------------
__global__ __launch_bounds__(256) void normalize_kernel(const float* __restrict__ F,
                                                        half_t* __restrict__ Vh) {
  int g = blockIdx.x * 256 + threadIdx.x;
  int b = g >> 12;            // / 4096
  int n = g & (N_ - 1);
  const float* col = F + (size_t)b * C_ * N_ + n;   // stride N_ between channels
  float ss = 0.f;
#pragma unroll 8
  for (int c = 0; c < C_; ++c) {
    float x = col[(size_t)c * N_];
    ss += x * x;
  }
  float inv = 1.f / fmaxf(sqrtf(ss), 1e-12f);
  half_t* dst = Vh + ((size_t)b * N_ + n) * C_;
#pragma unroll 8
  for (int c = 0; c < C_; ++c) dst[c] = (half_t)(col[(size_t)c * N_] * inv);
}

// ---------------------------------------------------------------------------
// Kernel 2: per-batch class histogram (denominators) + zero the output scalar.
// ---------------------------------------------------------------------------
__global__ __launch_bounds__(256) void hist_kernel(const int* __restrict__ L,
                                                   float* __restrict__ cnt,
                                                   float* __restrict__ out) {
  __shared__ int h[NCLS];
  int b = blockIdx.x;
  if (threadIdx.x < NCLS) h[threadIdx.x] = 0;
  __syncthreads();
  for (int n = threadIdx.x; n < N_; n += 256) atomicAdd(&h[L[b * N_ + n]], 1);
  __syncthreads();
  if (threadIdx.x < NCLS) cnt[b * NCLS + threadIdx.x] = (float)h[threadIdx.x];
  if (b == 0 && threadIdx.x == 0) out[0] = 0.f;
}

// ---------------------------------------------------------------------------
// Async stage of JB columns of Vh into LDS: each thread moves one 64B chunk
// of one column straight from global memory into LDS via the gfx1250
// GLOBAL_LOAD_ASYNC_TO_LDS_B128 path (ASYNCcnt, no VGPR round-trip).
// The 24-bit inst offset applies to BOTH the LDS and global address.
// ---------------------------------------------------------------------------
__device__ __forceinline__ void stage_cols_async(const half_t* __restrict__ Vbase,
                                                 half_t* dst, int jb, int tid) {
  const int col  = tid >> 2;
  const int part = (tid & 3) << 5;                 // 0,32,64,96 halves
  const half_t* src = Vbase + (size_t)(jb + col) * C_ + part;
  half_t* d = dst + col * LDS_STRIDE + part;
  unsigned lds_off = (unsigned)(size_t)d;          // flat LDS aperture: low 32 bits
  asm volatile(
      "global_load_async_to_lds_b128 %0, %1, off\n\t"
      "global_load_async_to_lds_b128 %0, %1, off offset:16\n\t"
      "global_load_async_to_lds_b128 %0, %1, off offset:32\n\t"
      "global_load_async_to_lds_b128 %0, %1, off offset:48"
      :: "v"(lds_off), "v"(src) : "memory");
}

__device__ __forceinline__ void wait_async_and_barrier() {
  asm volatile("s_wait_asynccnt 0x0" ::: "memory");
  __syncthreads();
}

// ---------------------------------------------------------------------------
// Kernel 3: Gram-row partial reduction via v_wmma_f32_16x16x32_f16.
// blockIdx.x = (b * JSPLIT + half) * 32 + g :
//   the 8 waves own i-tiles g*8..g*8+7 and sweep columns [half*JLEN, +JLEN).
// Partials (T = total expsum, S = same-class expsum) go to disjoint ws slots.
// ---------------------------------------------------------------------------
__global__ __launch_bounds__(256) void gram_kernel(const half_t* __restrict__ Vh,
                                                   const int* __restrict__ L,
                                                   float* __restrict__ Tws,
                                                   float* __restrict__ Sws) {
  __shared__ half_t bufS[2][JB * LDS_STRIDE];      // 2 x 17KB
  __shared__ int    labS[N_];                      // 16KB

  const int tid  = threadIdx.x;
  const int lane = tid & 31;
  const int m    = lane & 15;
  const int hi   = lane >> 4;

  // All of these are wave-uniform; force them scalar so per-tile control flow
  // is s_cmp/s_cbranch instead of EXEC-mask manipulation.
  const int wv    = __builtin_amdgcn_readfirstlane(tid >> 5);
  const int g     = blockIdx.x & 31;
  const int half  = (blockIdx.x >> 5) & 1;
  const int b     = blockIdx.x >> 6;
  const int i0    = (g * 8 + wv) << 4;             // this wave's row-tile base
  const int jbase = half * JLEN;

  const half_t* Vbase = Vh + (size_t)b * N_ * C_;

  // Stage labels + first column block before the first barrier.
  for (int n = tid; n < N_; n += 256) labS[n] = L[b * N_ + n];
  stage_cols_async(Vbase, bufS[0], jbase, tid);

  // --- A fragments: rows i0..i0+15, all K=0..127 (ISA 16-bit A 16x32 layout) ---
  // lane L: M = L&15; halves 0..7 -> K = kb + h      (kb = 32*kk + 8*hi)
  //                   halves 8..15 -> K = kb + 16 + h
  v16h a[4];
  {
    const half_t* rowp = Vbase + (size_t)(i0 + m) * C_;
#pragma unroll
    for (int kk = 0; kk < 4; ++kk) {
      int kb = kk * 32 + 8 * hi;
      v8h lo = *(const v8h*)(rowp + kb);
      v8h hv = *(const v8h*)(rowp + kb + 16);
      a[kk] = __builtin_shufflevector(lo, hv, 0, 1, 2, 3, 4, 5, 6, 7,
                                               8, 9, 10, 11, 12, 13, 14, 15);
    }
  }

  wait_async_and_barrier();

  int   li[8];
  float T[8], S[8];
#pragma unroll
  for (int r = 0; r < 8; ++r) {
    li[r] = labS[i0 + r + 8 * hi];    // C/D layout: VGPR r <-> row M = r + 8*hi
    T[r] = 0.f;
    S[r] = 0.f;
  }

  for (int blk = 0; blk < NBLK; ++blk) {
    const int cur = blk & 1;
    if (blk + 1 < NBLK)               // async-prefetch next block into other buffer
      stage_cols_async(Vbase, bufS[cur ^ 1], jbase + (blk + 1) * JB, tid);

    const half_t* base = bufS[cur];
#pragma unroll
    for (int t = 0; t < JB / 16; ++t) {
      const int j0 = jbase + blk * JB + t * 16;
      // B 32x16 layout: N = lane&15, K(h) = 8*hi + h (h<8), 16 + 8*hi + (h-8) (h>=8)
      const half_t* colp = base + (t * 16 + m) * LDS_STRIDE;

      // Load ALL four B fragments first (one dscnt wait), then 4 chained WMMAs.
      v16h bf[4];
#pragma unroll
      for (int kk = 0; kk < 4; ++kk) {
        int kb = kk * 32 + 8 * hi;
        v8h lo = *(const v8h*)(colp + kb);
        v8h hv = *(const v8h*)(colp + kb + 16);
        bf[kk] = __builtin_shufflevector(lo, hv, 0, 1, 2, 3, 4, 5, 6, 7,
                                                  8, 9, 10, 11, 12, 13, 14, 15);
      }
      v8f acc = {};
#pragma unroll
      for (int kk = 0; kk < 4; ++kk)
        acc = __builtin_amdgcn_wmma_f32_16x16x32_f16(false, a[kk], false, bf[kk],
                                                     (short)0, acc, false, false);

      const int lj = labS[j0 + m];
      if (j0 != i0) {                 // scalar-uniform branch: no diagonal here
#pragma unroll
        for (int r = 0; r < 8; ++r) {
          float e = __expf(acc[r]);   // TEMP cancels in pos/(pos+neg)
          T[r] += e;
          S[r] += (li[r] == lj) ? e : 0.f;
        }
      } else {                        // the single tile containing the diagonal
#pragma unroll
        for (int r = 0; r < 8; ++r) {
          float e = (r + 8 * hi == m) ? 0.f : __expf(acc[r]);
          T[r] += e;
          S[r] += (li[r] == lj) ? e : 0.f;
        }
      }
    }
    wait_async_and_barrier();
  }

  // Reduce across the 16 lanes that share a row (xor masks stay within halves).
#pragma unroll
  for (int r = 0; r < 8; ++r) {
#pragma unroll
    for (int off = 8; off >= 1; off >>= 1) {
      T[r] += __shfl_xor(T[r], off, 32);
      S[r] += __shfl_xor(S[r], off, 32);
    }
  }

  if (m == 0) {  // lanes 0 (rows 0..7) and 16 (rows 8..15)
#pragma unroll
    for (int r = 0; r < 8; ++r) {
      int row = i0 + r + 8 * hi;
      size_t slot = ((size_t)b * N_ + row) * JSPLIT + half;   // disjoint slots
      Tws[slot] = T[r];
      Sws[slot] = S[r];
    }
  }
}

// ---------------------------------------------------------------------------
// Kernel 4: combine j-split partials, per-row loss, global mean.
// ---------------------------------------------------------------------------
__global__ __launch_bounds__(256) void loss_kernel(const float* __restrict__ Tws,
                                                   const float* __restrict__ Sws,
                                                   const int* __restrict__ L,
                                                   const float* __restrict__ cnt,
                                                   float* __restrict__ out) {
  int g = blockIdx.x * 256 + threadIdx.x;          // g = b*N + n
  int b = g >> 12;
  float T = Tws[(size_t)g * JSPLIT] + Tws[(size_t)g * JSPLIT + 1];
  float S = Sws[(size_t)g * JSPLIT] + Sws[(size_t)g * JSPLIT + 1];
  int   li  = L[g];
  float c1  = cnt[b * NCLS + li];
  float pos = S / c1;                              // denom includes self (ref)
  float neg = (T - S) / ((float)N_ - c1);
  float loss = __logf((pos + neg) / pos);          // -log(pos/(pos+neg))
#pragma unroll
  for (int off = 16; off >= 1; off >>= 1) loss += __shfl_xor(loss, off, 32);
  if ((threadIdx.x & 31) == 0)
    atomicAdd(out, loss * (1.0f / (float)(B_ * N_)));
}

// ---------------------------------------------------------------------------
extern "C" void kernel_launch(void* const* d_in, const int* in_sizes, int n_in,
                              void* d_out, int out_size, void* d_ws, size_t ws_size,
                              hipStream_t stream) {
  (void)in_sizes; (void)n_in; (void)out_size; (void)ws_size;
  const float* F = (const float*)d_in[0];
  const int*   L = (const int*)d_in[1];
  float* out = (float*)d_out;

  char* ws = (char*)d_ws;
  half_t* Vh  = (half_t*)ws;                                   // 8 MB
  size_t off  = (size_t)B_ * N_ * C_ * sizeof(half_t);
  float*  cnt = (float*)(ws + off);  off += B_ * NCLS * sizeof(float);
  float*  Tws = (float*)(ws + off);  off += (size_t)B_ * N_ * JSPLIT * sizeof(float);
  float*  Sws = (float*)(ws + off);

  normalize_kernel<<<(B_ * N_) / 256, 256, 0, stream>>>(F, Vh);
  hist_kernel<<<B_, 256, 0, stream>>>(L, cnt, out);
  gram_kernel<<<B_ * JSPLIT * 32, 256, 0, stream>>>(Vh, L, Tws, Sws);
  loss_kernel<<<(B_ * N_) / 256, 256, 0, stream>>>(Tws, Sws, L, cnt, out);
}
